// AffinityDynamics_88562225643534
// MI455X (gfx1250) — matrix operand order repaired and checked by reading, not compiled, not run on hardware
//
#include <hip/hip_runtime.h>

// MI455X (gfx1250) — compute-bound fused attention.
//  - f16 WMMA (v_wmma_f32_16x16x32_f16) with f32 accumulation everywhere.
//  - Weights/tau pre-converted to f16 once; Q/K/V f16 workspace (48MB+2MB)
//    is L2-resident (192MB L2), so flash-attention K/V re-reads never hit HBM.
//  - Q block staged into LDS by the Tensor Data Mover (tensor_load_to_lds +
//    s_wait_tensorcnt) when the builtin is available.

typedef __attribute__((ext_vector_type(16))) _Float16 v16h;
typedef __attribute__((ext_vector_type(4)))  _Float16 v4h;
typedef __attribute__((ext_vector_type(8)))  float    v8f;
typedef __attribute__((ext_vector_type(4)))  unsigned int v4u;
typedef __attribute__((ext_vector_type(8)))  int      v8i;
typedef __attribute__((ext_vector_type(4)))  int      v4i;

#define WMMA_F16(a, b, c) \
  __builtin_amdgcn_wmma_f32_16x16x32_f16(false, (a), false, (b), (short)0, (c), false, false)

#if __has_builtin(__builtin_amdgcn_tensor_load_to_lds) && \
    __has_builtin(__builtin_amdgcn_s_wait_tensorcnt)
#define USE_TDM 1
#if __has_include(<hip/amd_detail/amd_gfx1250_TDM.h>)
// amdgpu-toolchain (clang-23 / therock-10.0): 6-arg form
#define TDM_LOAD(g0, g1, g2, g3) \
  __builtin_amdgcn_tensor_load_to_lds((g0), (g1), (g2), (g3), \
                                      (v8i){0, 0, 0, 0, 0, 0, 0, 0}, 0)
#else
// ROCm 7.2 (clang-22): 5-arg form
#define TDM_LOAD(g0, g1, g2, g3) \
  __builtin_amdgcn_tensor_load_to_lds((g0), (g1), (g2), (g3), 0)
#endif
#endif

// ---- A-operand fragment (16x32 f16, MxK) from a row-major f16 buffer ----
// Contiguous K layout: lane L<16 -> M=L, elements hold K=0..15
//               lane L>=16 -> M=L-16, elements hold K=16..31.
__device__ __forceinline__ v16h load_a_frag(const _Float16* base, int stride,
                                            int lane, int kbase) {
  v16h a;
  const int m   = lane & 15;
  const int hi16 = (lane >> 4) << 4;
  const _Float16* p = base + (size_t)m * stride + kbase;
#pragma unroll
  for (int i = 0; i < 16; ++i) {
    const int v = i >> 1;
    (void)v;
    a[i] = p[hi16 + i];
  }
  return a;
}

// =====================================================================
// Kernel 0: one-shot f32 -> f16 conversion of Wq|Wk|Wv and tau.
// =====================================================================
__global__ __launch_bounds__(256)
void cvt_f16_kernel(const float* __restrict__ Wq, const float* __restrict__ Wk,
                    const float* __restrict__ Wv, const float* __restrict__ tau,
                    _Float16* __restrict__ w16,   // wq|wk|wv, 3*512*512
                    _Float16* __restrict__ tau16) // 4096*64
{
  constexpr int W = 512 * 512, T = 4096 * 64;
  const int idx = blockIdx.x * blockDim.x + threadIdx.x;
  const int stride = gridDim.x * blockDim.x;
  for (int i = idx; i < W; i += stride) {
    w16[i]         = (_Float16)Wq[i];
    w16[W + i]     = (_Float16)Wk[i];
    w16[2 * W + i] = (_Float16)Wv[i];
  }
  for (int i = idx; i < T; i += stride) tau16[i] = (_Float16)tau[i];
}

// =====================================================================
// Kernel 1: fused Q/K/V = h @ W^T.  grid.x = B*N/16 row tiles, 8 waves.
// h block staged once through LDS as f16; each A-fragment feeds 12 WMMAs.
// =====================================================================
__global__ __launch_bounds__(256)
void qkv_gemm_kernel(const float* __restrict__ h,
                     const _Float16* __restrict__ w16,
                     _Float16* __restrict__ qkv16) {
  constexpr int D = 512;
  constexpr size_t MAT = (size_t)4 * 4096 * 512;
  __shared__ _Float16 hs[16 * 512];           // 16 KB
  const int rt   = blockIdx.x;
  const int lane = threadIdx.x & 31;
  const int wave = threadIdx.x >> 5;          // 0..7 -> 64-col chunk

  const float* hsrc = h + (size_t)rt * 16 * D;
  for (int i = threadIdx.x; i < 16 * D / 4; i += 256) {
    const float4 f = ((const float4*)hsrc)[i];
    v4h p = {(_Float16)f.x, (_Float16)f.y, (_Float16)f.z, (_Float16)f.w};
    ((v4h*)hs)[i] = p;
  }
  __syncthreads();

  const int m    = lane & 15;
  const int hi8  = (lane >> 4) << 3;
  const int hi16 = (lane >> 4) << 4;

  v8f acc[3][4] = {};
#pragma unroll 2
  for (int ks = 0; ks < D / 32; ++ks) {
    v16h a = load_a_frag(hs, D, lane, ks * 32);
#pragma unroll
    for (int wm = 0; wm < 3; ++wm)
#pragma unroll
      for (int ci = 0; ci < 4; ++ci) {
        const int e = wave * 64 + ci * 16 + m;
        const _Float16* wp = w16 + (size_t)wm * D * D + (size_t)e * D + ks * 32 + hi16;
        v16h bf;
#pragma unroll
        for (int i = 0; i < 16; ++i) bf[i] = wp[i];
        acc[wm][ci] = WMMA_F16(a, bf, acc[wm][ci]);
      }
  }
#pragma unroll
  for (int wm = 0; wm < 3; ++wm)
#pragma unroll
    for (int ci = 0; ci < 4; ++ci)
#pragma unroll
      for (int j = 0; j < 8; ++j) {
        const int row = rt * 16 + j + hi8;
        const int col = wave * 64 + ci * 16 + m;
        qkv16[(size_t)wm * MAT + (size_t)row * D + col] = (_Float16)acc[wm][ci][j];
      }
}

// =====================================================================
// Kernel 2: fused affinity + flash attention.
// grid = (N/32, B), 256 threads = 8 wave32s.
// LDS layout (single flat buffer so TDM lds_addr offsets are compile-time):
//   [0      .. 32768)  qs  : 32x512 f16 query block
//   [32768  .. 36864)  tq  : 32x64  f16 tau rows (queries)
//   [36864  .. 45056)  Sp  : 32x64  f32 scaled scores
//   [45056  .. 49152)  Pp  : 32x64  f16 softmax probs
//   [49152  .. 49536)  rowmax/rowsum/alphas
// =====================================================================
__global__ __launch_bounds__(256)
void affinity_attn_kernel(const _Float16* __restrict__ tau16,
                          const _Float16* __restrict__ qf16,
                          const _Float16* __restrict__ kf16,
                          const _Float16* __restrict__ vf16,
                          float* __restrict__ out) {
  constexpr int Nn = 4096, D = 512, R = 64;
  const int lane = threadIdx.x & 31;
  const int wave = threadIdx.x >> 5;
  const int n0   = blockIdx.x * 32;
  const int b    = blockIdx.y;

  __shared__ __align__(16) unsigned char smem[49536];
  _Float16* qs = (_Float16*)smem;
  _Float16* tq = (_Float16*)(smem + 32768);
  float*    Sp = (float*)(smem + 36864);
  _Float16* Pp = (_Float16*)(smem + 45056);
  float* rowmax = (float*)(smem + 49152);
  float* rowsum = rowmax + 32;
  float* alphas = rowsum + 32;

  // ---- stage Q block (TDM) + tau(query rows) ----
#ifdef USE_TDM
  {
    // One TDM descriptor per wave: 4 query rows (512 f16 each) -> LDS.
    const unsigned long long ga = (unsigned long long)(const void*)
        (qf16 + ((size_t)b * Nn + n0 + wave * 4) * D);
    v4u g0 = {1u,                                   // count=1 (valid user D#)
              (unsigned)(wave * 4 * D * 2),         // lds_addr (bytes)
              (unsigned)(ga & 0xFFFFFFFFu),         // global_addr[31:0]
              (unsigned)((ga >> 32) & 0x1FFFFFFu) | (2u << 30)}; // addr hi | type=2
    v8i g1 = {0x00010000,    // data_size=1 (2 bytes/elem)
              512 << 16,     // tensor_dim0[15:0]=512
              4 << 16,       // tensor_dim0 hi=0 | tensor_dim1[15:0]=4
              512 << 16,     // tensor_dim1 hi=0 | tile_dim0=512
              4,             // tile_dim1=4 | tile_dim2=0
              512,           // tensor_dim0_stride[31:0]=512
              0, 0};
    v4i gz = {0, 0, 0, 0};
    TDM_LOAD(g0, g1, gz, gz);
  }
#else
  {
    const uint4* qsrc = (const uint4*)(qf16 + ((size_t)b * Nn + n0) * D);
    uint4* qdst = (uint4*)qs;
    for (int i = threadIdx.x; i < 32 * 512 * 2 / 16; i += 256) qdst[i] = qsrc[i];
  }
#endif
  {
    const uint4* tsrc = (const uint4*)(tau16 + (size_t)n0 * R);
    uint4* tdst = (uint4*)tq;
    for (int i = threadIdx.x; i < 32 * 64 * 2 / 16; i += 256) tdst[i] = tsrc[i];
    if (threadIdx.x < 32) { rowmax[threadIdx.x] = -3.0e38f; rowsum[threadIdx.x] = 0.f; }
  }
#ifdef USE_TDM
  __builtin_amdgcn_s_wait_tensorcnt((short)0);
#endif
  __syncthreads();

  const int   m    = lane & 15;
  const int   hi8  = (lane >> 4) << 3;
  const int   hi16 = (lane >> 4) << 4;
  const float inv_scale = 0.04419417382415922f;   // 1/sqrt(512)
  const int   qi = wave >> 2;
  const int   kj = wave & 3;

  v8f oacc[2][4] = {};                            // 32 rows x 64 cols per wave

  for (int kt = 0; kt < Nn / 64; ++kt) {
    const int key_base = kt * 64;

    // prefetch next K/V tile rows (global_prefetch_b8; L2-resident data)
    if (kt + 1 < Nn / 64) {
      const size_t nrow = (size_t)b * Nn + key_base + 64 + (threadIdx.x >> 2);
      __builtin_prefetch(kf16 + nrow * D, 0, 1);
      __builtin_prefetch(vf16 + nrow * D, 0, 1);
    }

    // ---- S tile: Q K^T over D, plus tau tau^T over R ----
    v8f s = {};
    const int krow = key_base + kj * 16 + m;
    const _Float16* kp0 = kf16 + ((size_t)b * Nn + krow) * D + hi16;
#pragma unroll 4
    for (int ks = 0; ks < D / 32; ++ks) {
      v16h a = load_a_frag(qs + qi * 16 * 512, 512, lane, ks * 32);
      const _Float16* kp = kp0 + ks * 32;
      v16h bf;
#pragma unroll
      for (int i = 0; i < 16; ++i) bf[i] = kp[i];
      s = WMMA_F16(a, bf, s);
    }
#pragma unroll
    for (int ks = 0; ks < R / 32; ++ks) {
      v16h a = load_a_frag(tq + qi * 16 * 64, 64, lane, ks * 32);
      const _Float16* tp = tau16 + (size_t)krow * R + ks * 32 + hi16;
      v16h bf;
#pragma unroll
      for (int i = 0; i < 16; ++i) bf[i] = tp[i];
      s = WMMA_F16(a, bf, s);
    }
#pragma unroll
    for (int j = 0; j < 8; ++j)
      Sp[(qi * 16 + j + hi8) * 64 + kj * 16 + m] = s[j] * inv_scale;
    __syncthreads();

    // ---- online softmax: 8 lanes per row, shfl_xor reductions ----
    {
      const int r  = threadIdx.x >> 3;        // 0..31
      const int c0 = (threadIdx.x & 7) << 3;  // 0..56
      float sv[8];
      float mloc = -3.0e38f;
#pragma unroll
      for (int c = 0; c < 8; ++c) { sv[c] = Sp[r * 64 + c0 + c]; mloc = fmaxf(mloc, sv[c]); }
#pragma unroll
      for (int off = 1; off < 8; off <<= 1) mloc = fmaxf(mloc, __shfl_xor(mloc, off, 8));
      const float mold = rowmax[r];
      const float mnew = fmaxf(mold, mloc);
      float sloc = 0.f;
#pragma unroll
      for (int c = 0; c < 8; ++c) {
        const float p = __expf(sv[c] - mnew);
        Pp[r * 64 + c0 + c] = (_Float16)p;
        sloc += p;
      }
#pragma unroll
      for (int off = 1; off < 8; off <<= 1) sloc += __shfl_xor(sloc, off, 8);
      if ((threadIdx.x & 7) == 0) {
        const float al = __expf(mold - mnew);
        rowsum[r] = rowsum[r] * al + sloc;
        rowmax[r] = mnew;
        alphas[r] = al;
      }
    }
    __syncthreads();

    // ---- rescale O, then O += P V for this wave's 64-col chunk ----
#pragma unroll
    for (int q2 = 0; q2 < 2; ++q2) {
      float aj[8];
#pragma unroll
      for (int j = 0; j < 8; ++j) aj[j] = alphas[q2 * 16 + j + hi8];
#pragma unroll
      for (int ci = 0; ci < 4; ++ci)
#pragma unroll
        for (int j = 0; j < 8; ++j) oacc[q2][ci][j] *= aj[j];
    }
#pragma unroll
    for (int ks = 0; ks < 2; ++ks) {
      v16h pa0 = load_a_frag(Pp,           64, lane, ks * 32);
      v16h pa1 = load_a_frag(Pp + 16 * 64, 64, lane, ks * 32);
      const size_t vbase =
          ((size_t)b * Nn + key_base + ks * 32 + hi16) * D + wave * 64 + m;
#pragma unroll
      for (int ci = 0; ci < 4; ++ci) {
        v16h bf;
#pragma unroll
        for (int i = 0; i < 16; ++i) bf[i] = vf16[vbase + (size_t)i * D + ci * 16];
        oacc[0][ci] = WMMA_F16(pa0, bf, oacc[0][ci]);
        oacc[1][ci] = WMMA_F16(pa1, bf, oacc[1][ci]);
      }
    }
    __syncthreads();
  }

  // ---- normalize and store f32 output ----
#pragma unroll
  for (int q2 = 0; q2 < 2; ++q2)
#pragma unroll
    for (int ci = 0; ci < 4; ++ci)
#pragma unroll
      for (int j = 0; j < 8; ++j) {
        const int lr  = q2 * 16 + j + hi8;
        const int row = n0 + lr;
        const int col = wave * 64 + ci * 16 + m;
        out[((size_t)b * Nn + row) * D + col] = oacc[q2][ci][j] / rowsum[lr];
      }
}

extern "C" void kernel_launch(void* const* d_in, const int* in_sizes, int n_in,
                              void* d_out, int out_size, void* d_ws, size_t ws_size,
                              hipStream_t stream) {
  // setup_inputs order: t, h, Wq, Wk, Wv, tau
  const float* h   = (const float*)d_in[1];
  const float* Wq  = (const float*)d_in[2];
  const float* Wk  = (const float*)d_in[3];
  const float* Wv  = (const float*)d_in[4];
  const float* tau = (const float*)d_in[5];
  float* out = (float*)d_out;

  constexpr int B = 4, N = 4096, D = 512;
  constexpr size_t MAT = (size_t)B * N * D;

  _Float16* qkv16 = (_Float16*)d_ws;                  // 48 MB (q|k|v)
  _Float16* w16   = qkv16 + 3 * MAT;                  // 1.5 MB (wq|wk|wv)
  _Float16* tau16 = w16 + (size_t)3 * D * D;          // 0.5 MB

  cvt_f16_kernel<<<dim3(512), 256, 0, stream>>>(Wq, Wk, Wv, tau, w16, tau16);
  qkv_gemm_kernel<<<dim3((B * N) / 16), 256, 0, stream>>>(h, w16, qkv16);
  affinity_attn_kernel<<<dim3(N / 32, B), 256, 0, stream>>>(
      tau16, qkv16, qkv16 + MAT, qkv16 + 2 * MAT, out);
}
